// ProcessorBlock_62758062129887
// MI455X (gfx1250) — matrix operand order repaired and checked by reading, not compiled, not run on hardware
//
#include <hip/hip_runtime.h>
#include <hip/hip_bf16.h>
#include <math.h>

#define NN   32768
#define EE   131072
#define HID  256
#define HEADS 8
#define CH   32

typedef __attribute__((ext_vector_type(16))) __bf16 v16bf;
typedef __attribute__((ext_vector_type(8)))  float  v8f;

union FragBF { unsigned int u[8]; v16bf v; };

__device__ __forceinline__ unsigned short f2bf(float f) {
  unsigned u = __float_as_uint(f);
  u += 0x7FFFu + ((u >> 16) & 1u);       // round-to-nearest-even truncation
  return (unsigned short)(u >> 16);
}
__device__ __forceinline__ unsigned f2ord(float f) {   // order-preserving float->uint
  unsigned u = __float_as_uint(f);
  return (u & 0x80000000u) ? ~u : (u | 0x80000000u);
}
__device__ __forceinline__ float ord2f(unsigned u) {
  return (u & 0x80000000u) ? __uint_as_float(u ^ 0x80000000u) : __uint_as_float(~u);
}
__device__ __forceinline__ float sigmoidf(float x) { return 1.0f / (1.0f + expf(-x)); }
__device__ __forceinline__ float eluf(float x) { return x > 0.0f ? x : expm1f(x); }

// ---------------------------------------------------------------------------
// GEMM: Y[NN,256] = epi( X[NN,256] @ W[256,256] + bias (+Y if ACCUM) )
// bf16 WMMA 16x16x32, f32 accumulate.
// Block: 256 threads (8 waves). Block tile 128 rows x 64 cols.
// Wave w: rows [16w,16w+16), 4 col tiles of 16.
// EPI: 0 = identity, 1 = tanh, 2 = elu(y + extra[row,col])
// ---------------------------------------------------------------------------
#define AS_STRIDE 72    // ushorts: K-chunk of 64 + 8 pad (conflict-free lane stride)
#define BS_STRIDE 264   // ushorts: K=256 + 8 pad

template <bool ACCUM, int EPI>
__global__ __launch_bounds__(256)
void gemm_wmma_kernel(const float* __restrict__ X, const float* __restrict__ W,
                      const float* __restrict__ bias, float* __restrict__ Y,
                      const float* __restrict__ extra)
{
  __shared__ unsigned short Bs[64 * BS_STRIDE];   // [n][k]   33792 B
  __shared__ unsigned short As[128 * AS_STRIDE];  // [row][k chunk] 18432 B

  const int rowblk = blockIdx.x * 128;
  const int colblk = blockIdx.y * 64;
  const int tid  = threadIdx.x;
  const int wid  = tid >> 5;
  const int lane = tid & 31;
  const int m    = lane & 15;
  const int khalf = lane >> 4;
  const int mrow = wid * 16;

  // Load W tile 256(K) x 64(N), store transposed (n-major, K contiguous), f32->bf16
  for (int i = tid; i < 64 * 64; i += 256) {     // 4096 float4s
    int k  = i >> 4;
    int n4 = (i & 15) << 2;
    float4 f = *(const float4*)(W + k * HID + colblk + n4);
    Bs[(n4 + 0) * BS_STRIDE + k] = f2bf(f.x);
    Bs[(n4 + 1) * BS_STRIDE + k] = f2bf(f.y);
    Bs[(n4 + 2) * BS_STRIDE + k] = f2bf(f.z);
    Bs[(n4 + 3) * BS_STRIDE + k] = f2bf(f.w);
  }

  v8f acc[4];
  #pragma unroll
  for (int t = 0; t < 4; ++t)
    #pragma unroll
    for (int j = 0; j < 8; ++j) acc[t][j] = 0.0f;

  for (int kc = 0; kc < 4; ++kc) {               // K chunks of 64
    const int kbase = kc * 64;
    __syncthreads();                              // protect As reuse
    for (int i = tid; i < 128 * 16; i += 256) {   // 2048 float4s
      int r  = i >> 4;
      int c4 = (i & 15) << 2;
      float4 f = *(const float4*)(X + (size_t)(rowblk + r) * HID + kbase + c4);
      unsigned short* d = &As[r * AS_STRIDE + c4];
      d[0] = f2bf(f.x); d[1] = f2bf(f.y); d[2] = f2bf(f.z); d[3] = f2bf(f.w);
    }
    __syncthreads();

    #pragma unroll
    for (int k0i = 0; k0i < 2; ++k0i) {           // two 32-deep WMMA steps
      const int k0 = k0i * 32;
      // A fragment: lane = row M (l&15), lane half selects K sub-block
      FragBF a;
      const unsigned short* arow = &As[(mrow + m) * AS_STRIDE + k0 + khalf * 8];
      #pragma unroll
      for (int j = 0; j < 8; ++j) {
        int off = ((j & 3) << 1) + ((j >> 2) << 4);   // K pairs 0..7 then 16..23
        a.u[j] = *(const unsigned int*)(arow + off);
      }
      #pragma unroll
      for (int t = 0; t < 4; ++t) {
        // B fragment: lane = col N, elems = K (same order as A: 8h..8h+7, 16+8h..16+8h+7)
        FragBF b;
        const unsigned short* brow =
            &Bs[(t * 16 + m) * BS_STRIDE + kbase + k0 + khalf * 8];
        #pragma unroll
        for (int j = 0; j < 8; ++j)
          b.u[j] = *(const unsigned int*)(brow + ((j & 3) << 1) + ((j >> 2) << 4));
        acc[t] = __builtin_amdgcn_wmma_f32_16x16x32_bf16(
            false, a.v, false, b.v, (short)0, acc[t], false, false);
      }
    }
  }

  // Store: C/D layout -> VGPR r, lanes 0-15: M=r, lanes 16-31: M=r+8
  #pragma unroll
  for (int t = 0; t < 4; ++t) {
    const int col = colblk + t * 16 + m;
    const float bval = bias[col];
    #pragma unroll
    for (int r = 0; r < 8; ++r) {
      const size_t row = rowblk + mrow + r + khalf * 8;
      float val = acc[t][r] + bval;
      if (ACCUM) val += Y[row * HID + col];
      if (EPI == 1) val = tanhf(val);
      if (EPI == 2) val = eluf(val + extra[row * HID + col]);
      Y[row * HID + col] = val;
    }
  }
}

// ---------------------------------------------------------------------------
// Elementwise / edge kernels
// ---------------------------------------------------------------------------
__global__ __launch_bounds__(256)
void fill_u32_kernel(unsigned* p, unsigned v, int n) {
  int t = blockIdx.x * 256 + threadIdx.x;
  if (t < n) p[t] = v;
}

__global__ __launch_bounds__(256)
void embed_kernel(const float* __restrict__ x, const float* __restrict__ nd,
                  const float* __restrict__ Wnd, const float* __restrict__ bnd,
                  float* __restrict__ out) {
  int t = blockIdx.x * 256 + threadIdx.x;
  int n = t >> 8, c = t & 255;
  out[t] = x[t] + nd[n * 3 + 0] * Wnd[c] + nd[n * 3 + 1] * Wnd[HID + c]
               + nd[n * 3 + 2] * Wnd[2 * HID + c] + bnd[c];
}

__global__ void wesum_kernel(const float* __restrict__ We, float* __restrict__ wesum) {
  int t = threadIdx.x;                  // 32 threads
  int d = t >> 3, h = t & 7;
  float s = 0.0f;
  for (int c = 0; c < CH; ++c) s += We[d * HID + h * CH + c];
  wesum[d * HEADS + h] = s;             // sum over channels of eproj per head
}

__global__ __launch_bounds__(256)
void edge_attn_kernel(const float* __restrict__ q, const float* __restrict__ k,
                      const float* __restrict__ edata, const int* __restrict__ src,
                      const int* __restrict__ dst, const float* __restrict__ wesum,
                      float* __restrict__ attn, unsigned* __restrict__ m_ord) {
  int t = blockIdx.x * 256 + threadIdx.x;
  if (t >= EE * HEADS) return;
  int e = t >> 3, h = t & 7;
  int s = src[e], d = dst[e];
  const float4* kp = (const float4*)(k + (size_t)s * HID + h * CH);
  const float4* qp = (const float4*)(q + (size_t)d * HID + h * CH);
  float acc = 0.0f;
  #pragma unroll
  for (int i = 0; i < 8; ++i) {
    float4 a = kp[i], b = qp[i];
    acc += a.x * b.x + a.y * b.y + a.z * b.z + a.w * b.w;
  }
  float val = acc * 0.17677669529663687f;  // 1/sqrt(32)
  #pragma unroll
  for (int dd = 0; dd < 4; ++dd) val += edata[e * 4 + dd] * wesum[dd * HEADS + h];
  attn[t] = val;
  atomicMax(&m_ord[(size_t)d * HEADS + h], f2ord(val));
}

__global__ __launch_bounds__(256)
void edge_exp_kernel(const int* __restrict__ dst, const unsigned* __restrict__ m_ord,
                     float* __restrict__ attn, float* __restrict__ s_sum) {
  int t = blockIdx.x * 256 + threadIdx.x;
  if (t >= EE * HEADS) return;
  int e = t >> 3, h = t & 7;
  int d = dst[e];
  float ex = expf(attn[t] - ord2f(m_ord[(size_t)d * HEADS + h]));
  attn[t] = ex;
  atomicAdd(&s_sum[(size_t)d * HEADS + h], ex);
}

__global__ __launch_bounds__(256)
void edge_agg_kernel(const float* __restrict__ v, const int* __restrict__ src,
                     const int* __restrict__ dst, const float* __restrict__ attn,
                     const float* __restrict__ s_sum, float* __restrict__ agg) {
  int e = blockIdx.x;
  int col = threadIdx.x;
  int h = col >> 5;
  int s = src[e], d = dst[e];
  float w = attn[(size_t)e * HEADS + h] / s_sum[(size_t)d * HEADS + h];
  atomicAdd(&agg[(size_t)d * HID + col], w * v[(size_t)s * HID + col]);
}

__global__ __launch_bounds__(256)
void ew_rh_kernel(float* __restrict__ r, const float* __restrict__ h) {
  int t = blockIdx.x * 256 + threadIdx.x;
  r[t] = sigmoidf(r[t]) * h[t];
}

__global__ __launch_bounds__(256)
void ew_gate_kernel(float* __restrict__ H, const float* __restrict__ zi,
                    const float* __restrict__ hh) {
  int t = blockIdx.x * 256 + threadIdx.x;
  float z = sigmoidf(zi[t]);
  H[t] = (1.0f - z) * H[t] + z * tanhf(hh[t]);
}

__global__ __launch_bounds__(256)
void norm_stat_kernel(const float* __restrict__ x, float* __restrict__ sum_ws,
                      float* __restrict__ sq_ws) {
  int col = threadIdx.x;
  int rb = blockIdx.x * 128;
  float s = 0.0f, q = 0.0f;
  for (int r = 0; r < 128; ++r) {
    float v = x[(size_t)(rb + r) * HID + col];
    s += v; q += v * v;
  }
  atomicAdd(&sum_ws[col], s);
  atomicAdd(&sq_ws[col], q);
}

__global__ __launch_bounds__(256)
void norm_apply_elu_kernel(float* __restrict__ out, const float* __restrict__ x,
                           const float* __restrict__ sum_ws, const float* __restrict__ sq_ws,
                           const float* __restrict__ lnw, const float* __restrict__ lnb) {
  int t = blockIdx.x * 256 + threadIdx.x;
  int c = t & 255;
  const float inv = 1.0f / (float)NN;
  float mean = sum_ws[c] * inv;
  float var = fmaxf(sq_ws[c] * inv - mean * mean, 0.0f);
  float y = (x[t] - mean) / (sqrtf(var) + 1e-5f) * lnw[c] + lnb[c];
  out[t] = eluf(y);
}

// ---------------------------------------------------------------------------
// Host orchestration
// ---------------------------------------------------------------------------
extern "C" void kernel_launch(void* const* d_in, const int* in_sizes, int n_in,
                              void* d_out, int out_size, void* d_ws, size_t ws_size,
                              hipStream_t stream) {
  const float* x_in  = (const float*)d_in[0];
  const float* edata = (const float*)d_in[1];
  const float* ndata = (const float*)d_in[2];
  const int*   src   = (const int*)d_in[3];
  const int*   dst   = (const int*)d_in[4];
  // params, recursive insertion-order flatten:
  // conv[i] base = 5 + 9*i : Wq bq Wk bk Wv bv We Wr br
  // gate base = 23: Wzx bzx Wzh bzh Wrx brx Wrh brh Whx bhx Whh bhh
  // 35 ln_w, 36 ln_b, 37 Wn, 38 bn, 39 Wnd, 40 bnd, 41 Wo, 42 bo
  auto F = [&](int i) { return (const float*)d_in[i]; };

  const size_t BUF = (size_t)NN * HID;
  float* X  = (float*)d_ws;
  float* Qb = X + BUF;
  float* Kb = Qb + BUF;
  float* Vb = Kb + BUF;
  float* AG = Vb + BUF;
  float* Hb = AG + BUF;
  float* attn = Hb + BUF;                                   // EE*HEADS
  unsigned* m_ord = (unsigned*)(attn + (size_t)EE * HEADS); // NN*HEADS
  float* s_sum = (float*)(m_ord + (size_t)NN * HEADS);      // NN*HEADS
  float* wesum = s_sum + (size_t)NN * HEADS;                // 32
  float* stat_sum = wesum + 32;                             // 256
  float* stat_sq  = stat_sum + 256;                         // 256

  const dim3 ggrid(NN / 128, HID / 64);
  const int EWG = (NN * HID) / 256;       // 32768 blocks for N*HID elementwise
  const int EHG = (EE * HEADS) / 256;     // 4096 blocks for edge*head
  const int NHG = (NN * HEADS + 255) / 256;

  auto gemm = [&](const float* A, const float* W, const float* b, float* Y) {
    gemm_wmma_kernel<false, 0><<<ggrid, 256, 0, stream>>>(A, W, b, Y, nullptr);
  };
  auto gemm_acc = [&](const float* A, const float* W, const float* b, float* Y) {
    gemm_wmma_kernel<true, 0><<<ggrid, 256, 0, stream>>>(A, W, b, Y, nullptr);
  };
  auto gemm_tanh = [&](const float* A, const float* W, const float* b, float* Y) {
    gemm_wmma_kernel<false, 1><<<ggrid, 256, 0, stream>>>(A, W, b, Y, nullptr);
  };
  auto gemm_add_elu = [&](const float* A, const float* W, const float* b, float* Y,
                          const float* extra) {
    gemm_wmma_kernel<false, 2><<<ggrid, 256, 0, stream>>>(A, W, b, Y, extra);
  };

  // x = x_in + ndata @ Wnd + bnd
  embed_kernel<<<EWG, 256, 0, stream>>>(x_in, ndata, F(39), F(40), X);

  for (int layer = 0; layer < 2; ++layer) {
    const int cb = 5 + 9 * layer;
    // projections
    gemm(X, F(cb + 0), F(cb + 1), Qb);   // q
    gemm(X, F(cb + 2), F(cb + 3), Kb);   // k
    gemm(X, F(cb + 4), F(cb + 5), Vb);   // v
    gemm(X, F(cb + 7), F(cb + 8), AG);   // root lin (aggregation target)
    wesum_kernel<<<1, 32, 0, stream>>>(F(cb + 6), wesum);
    // edge softmax + aggregate
    fill_u32_kernel<<<NHG, 256, 0, stream>>>(m_ord, 0u, NN * HEADS);
    fill_u32_kernel<<<NHG, 256, 0, stream>>>((unsigned*)s_sum, 0u, NN * HEADS);
    edge_attn_kernel<<<EHG, 256, 0, stream>>>(Qb, Kb, edata, src, dst, wesum, attn, m_ord);
    edge_exp_kernel<<<EHG, 256, 0, stream>>>(dst, m_ord, attn, s_sum);
    edge_agg_kernel<<<EE, 256, 0, stream>>>(Vb, src, dst, attn, s_sum, AG);
    // gate
    if (layer == 0) {
      gemm_tanh(AG, F(31), F(32), Hb);                   // h = tanh(x@Whx+bhx)
      // graph norm + elu + linear -> next x
      fill_u32_kernel<<<2, 256, 0, stream>>>((unsigned*)stat_sum, 0u, 512);
      norm_stat_kernel<<<NN / 128, 256, 0, stream>>>(AG, stat_sum, stat_sq);
      norm_apply_elu_kernel<<<EWG, 256, 0, stream>>>(Qb, AG, stat_sum, stat_sq,
                                                     F(35), F(36));
      gemm(Qb, F(37), F(38), X);                         // x = elu(norm) @ Wn + bn
    } else {
      gemm(AG, F(23), F(24), Qb);                        // zi  = x@Wzx+bzx
      gemm_acc(Hb, F(25), F(26), Qb);                    //    += h@Wzh+bzh
      gemm(AG, F(27), F(28), Kb);                        // ri  = x@Wrx+brx
      gemm_acc(Hb, F(29), F(30), Kb);                    //    += h@Wrh+brh
      ew_rh_kernel<<<EWG, 256, 0, stream>>>(Kb, Hb);     // rh = sigmoid(ri)*h
      gemm(Kb, F(33), F(34), Vb);                        // hh_pre = rh@Whh+bhh
      ew_gate_kernel<<<EWG, 256, 0, stream>>>(Hb, Qb, Vb); // h=(1-z)h+z*tanh(hh)
    }
  }

  // out = elu(h @ Wo + bo + x_skip), fused epilogue writes d_out directly
  gemm_add_elu(Hb, F(41), F(42), (float*)d_out, x_in);
}
